// YOLO_71975061946924
// MI455X (gfx1250) — matrix operand order repaired and checked by reading, not compiled, not run on hardware
//
#include <hip/hip_runtime.h>

typedef __attribute__((ext_vector_type(16))) _Float16 v16h;
typedef __attribute__((ext_vector_type(8)))  float    v8f;

#define FMP    19
#define NPIX   361            // 19*19
#define PW     21             // padded width (halo of 1)
#define PPIX   441            // 21*21
#define NB     16
#define CH     512
#define G_TOT  (NB * NPIX)    // 5776
#define NCLS   20
#define CONF_T 0.001f
#define NMS_T  0.6f
#define SLOPE_C 0.1f

__device__ __forceinline__ int gpad_of(int b, int y, int x) {
  return b * PPIX + (y + 1) * PW + (x + 1);
}

// ---------------- zero padded activation buffers ----------------
__global__ void k_zero(uint4* __restrict__ p, int n16) {
  int i = blockIdx.x * blockDim.x + threadIdx.x;
  if (i < n16) p[i] = make_uint4(0u, 0u, 0u, 0u);
}

// ---------------- layout converts ----------------

// x: f32 NCHW [16][512][361]  ->  act: f16 padded NHWC [16][21][21][512] interior
__global__ void k_convert_x(const float* __restrict__ x, _Float16* __restrict__ act) {
  int o = blockIdx.x * blockDim.x + threadIdx.x;
  if (o >= G_TOT * CH) return;
  int g = o >> 9;          // /512
  int c = o & 511;
  int b = g / NPIX;
  int p = g - b * NPIX;
  int y = p / FMP;
  int xx = p - y * FMP;
  act[(size_t)gpad_of(b, y, xx) * CH + c] =
      (_Float16)x[((size_t)b * CH + c) * NPIX + p];
}

// w: f32 OIHW [512][512][3][3] -> f16 [r=9][oc=512][ic=512]
__global__ void k_convert_w(const float* __restrict__ w, _Float16* __restrict__ wh) {
  int n = blockIdx.x * blockDim.x + threadIdx.x;
  if (n >= CH * CH * 9) return;
  int oc  = n / (CH * 9);
  int rem = n - oc * (CH * 9);
  int ic  = rem / 9;
  int r   = rem - ic * 9;
  wh[((size_t)r * CH + oc) * CH + ic] = (_Float16)w[n];
}

// ---------------- 3x3 conv as 9 shifted GEMMs via WMMA ----------------
// One wave computes a 16(oc) x 16(pixel) f32 tile: 9 taps * 16 k-tiles = 144 WMMAs.
// Padded-NHWC input makes every B-fragment load unconditional (halo is zero).
__global__ __launch_bounds__(128) void k_conv3x3_wmma(
    const _Float16* __restrict__ actIn,   // [16*441][512] padded NHWC half
    const _Float16* __restrict__ wHalf,   // [9][512 oc][512 ic] half
    const float*    __restrict__ bias,    // [512]
    _Float16*       __restrict__ actOut)  // [16*441][512] padded NHWC half
{
  const int wave = threadIdx.x >> 5;
  const int lane = threadIdx.x & 31;
  const int ocTile  = blockIdx.x * 4 + wave;   // 0..31
  const int pixTile = blockIdx.y;              // 0..360

  const int col = lane & 15;     // N (pixel) column / A row
  const int hi  = lane >> 4;     // lane half

  const int g = pixTile * 16 + col;            // 0..5775 (exact)
  const int b = g / NPIX;
  const int p = g - b * NPIX;
  const int y = p / FMP;
  const int x = p - y * FMP;

  v8f acc = {};

  const int arow = col;   // A-matrix row (M = lane%16)

  for (int r = 0; r < 9; ++r) {
    const int dy = r / 3 - 1;
    const int dx = r - (r / 3) * 3 - 1;
    // always in-bounds thanks to zero halo
    const _Float16* bbase = actIn + ((size_t)(b * PPIX + (y + 1 + dy) * PW + (x + 1 + dx))) * CH;
    const _Float16* abase = wHalf + ((size_t)r * CH + (ocTile * 16 + arow)) * CH;

    if (r < 8) {  // prefetch next tap's activation row (global_prefetch_b8)
      const int r2 = r + 1;
      const int dy2 = r2 / 3 - 1;
      const int dx2 = r2 - (r2 / 3) * 3 - 1;
      __builtin_prefetch(actIn + ((size_t)(b * PPIX + (y + 1 + dy2) * PW + (x + 1 + dx2))) * CH, 0, 3);
    }

    for (int kt = 0; kt < 16; ++kt) {
      const int k0 = kt * 32;

      // A: 16-bit 16x32 layout — lane half 'hi' holds K runs [hi*8,hi*8+8) and [16+hi*8,...)
      v16h afrag;
      ((uint4*)&afrag)[0] = *(const uint4*)(abase + k0 + hi * 8);
      ((uint4*)&afrag)[1] = *(const uint4*)(abase + k0 + 16 + hi * 8);

      // B: 16-bit 32x16 layout — lane half 'hi' holds K runs [hi*8,hi*8+8) and [16+hi*8,...)
      v16h bfrag;
      ((uint4*)&bfrag)[0] = *(const uint4*)(bbase + k0 + hi * 8);
      ((uint4*)&bfrag)[1] = *(const uint4*)(bbase + k0 + 16 + hi * 8);

      acc = __builtin_amdgcn_wmma_f32_16x16x32_f16(
          /*neg_a=*/false, afrag, /*neg_b=*/false, bfrag,
          /*c_mod=*/(short)0, acc, /*reuse_a=*/false, /*reuse_b=*/false);
    }
  }

  // Epilogue: lane holds D for pixel g, oc = ocTile*16 + hi*8 + i (8 consecutive oc)
  const int ocBase = ocTile * 16 + hi * 8;
  union { _Float16 h[8]; uint4 u[1]; } o;
  #pragma unroll
  for (int i = 0; i < 8; ++i) {
    float v = acc[i] + bias[ocBase + i];
    v = v > 0.0f ? v : SLOPE_C * v;
    o.h[i] = (_Float16)v;
  }
  *(uint4*)(actOut + (size_t)gpad_of(b, y, x) * CH + ocBase) = o.u[0];
}

// ---------------- 1x1 heads ----------------
// heads layout per pixel: [0]=obj, [1..20]=cls logits, [21..24]=reg
__global__ void k_heads(const _Float16* __restrict__ cf, const _Float16* __restrict__ rf,
                        const float* __restrict__ wObj, const float* __restrict__ bObj,
                        const float* __restrict__ wCls, const float* __restrict__ bCls,
                        const float* __restrict__ wReg, const float* __restrict__ bReg,
                        float* __restrict__ heads) {
  int t = blockIdx.x * blockDim.x + threadIdx.x;
  if (t >= G_TOT * 25) return;
  int g = t / 25;
  int j = t - g * 25;
  int b = g / NPIX;
  int p = g - b * NPIX;
  int y = p / FMP;
  int xx = p - y * FMP;
  const size_t abase = (size_t)gpad_of(b, y, xx) * CH;
  const _Float16* act;
  const float* wrow;
  float bv;
  if (j == 0)      { act = rf; wrow = wObj;                 bv = bObj[0]; }
  else if (j < 21) { act = cf; wrow = wCls + (j - 1) * CH;  bv = bCls[j - 1]; }
  else             { act = rf; wrow = wReg + (j - 21) * CH; bv = bReg[j - 21]; }
  const _Float16* a = act + abase;
  float s = bv;
  for (int c = 0; c < CH; ++c) s += (float)a[c] * wrow[c];
  heads[(size_t)g * 25 + j] = s;
}

// ---------------- decode boxes + scores ----------------
// d_out: boxes[G*4] | sc[G] | cls[G] | keep[G]
__global__ void k_decode(const float* __restrict__ heads, float* __restrict__ dout) {
  int g = blockIdx.x * blockDim.x + threadIdx.x;
  if (g >= G_TOT) return;
  int b = g / NPIX;
  int p = g - b * NPIX;
  int gy = p / FMP;
  int gx = p - gy * FMP;
  const float* h = heads + (size_t)g * 25;

  float obj = h[0];
  float m = h[1];
  for (int c = 1; c < NCLS; ++c) m = fmaxf(m, h[1 + c]);
  float e[NCLS];
  float sum = 0.0f;
  for (int c = 0; c < NCLS; ++c) { e[c] = expf(h[1 + c] - m); sum += e[c]; }
  float objs = 1.0f / (1.0f + expf(-obj));
  float inv = 1.0f / sum;
  float best = -3.4e38f;
  int bi = 0;
  for (int c = 0; c < NCLS; ++c) {
    float scv = objs * e[c] * inv;
    if (scv > best) { best = scv; bi = c; }   // first-max, like jnp.argmax
  }

  float cx = 1.0f / (1.0f + expf(-h[21])) + (float)gx;
  float cy = 1.0f / (1.0f + expf(-h[22])) + (float)gy;
  float w2 = 0.5f * expf(h[23]);
  float h2 = 0.5f * expf(h[24]);
  const float scale = 32.0f / 608.0f;
  float x1 = fminf(fmaxf((cx - w2) * scale, 0.0f), 1.0f);
  float y1 = fminf(fmaxf((cy - h2) * scale, 0.0f), 1.0f);
  float x2 = fminf(fmaxf((cx + w2) * scale, 0.0f), 1.0f);
  float y2 = fminf(fmaxf((cy + h2) * scale, 0.0f), 1.0f);

  float* bx = dout + (size_t)g * 4;
  bx[0] = x1; bx[1] = y1; bx[2] = x2; bx[3] = y2;
  dout[(size_t)G_TOT * 4 + g] = best;
  dout[(size_t)G_TOT * 5 + g] = (float)bi;
  dout[(size_t)G_TOT * 6 + g] = 0.0f;   // keep init each call
}

// ---------------- per-(class,batch) NMS ----------------
__global__ __launch_bounds__(384) void k_nms(const float* __restrict__ dout,
                                             float* __restrict__ keepOut) {
  const int c = blockIdx.x;   // class
  const int b = blockIdx.y;   // batch
  const int tid = threadIdx.x;

  __shared__ float sRaw[NPIX];
  __shared__ float sS[NPIX];
  __shared__ float sX1[NPIX], sY1[NPIX], sX2[NPIX], sY2[NPIX], sA[NPIX];
  __shared__ int   sIdx[NPIX];
  __shared__ int   sKeep[NPIX];

  const float* boxes = dout;
  const float* sc    = dout + (size_t)G_TOT * 4;
  const float* cls   = dout + (size_t)G_TOT * 5;

  float myS = -1.0f, bx1 = 0, by1 = 0, bx2 = 0, by2 = 0;
  if (tid < NPIX) {
    int g = b * NPIX + tid;
    float s = sc[g];
    int ci = (int)cls[g];
    myS = (ci == c && s >= CONF_T) ? s : -1.0f;
    sRaw[tid] = myS;
    bx1 = boxes[(size_t)g * 4 + 0]; by1 = boxes[(size_t)g * 4 + 1];
    bx2 = boxes[(size_t)g * 4 + 2]; by2 = boxes[(size_t)g * 4 + 3];
  }
  __syncthreads();

  // stable rank sort (descending score, ties -> lower index first)
  if (tid < NPIX) {
    int rank = 0;
    for (int j = 0; j < NPIX; ++j) {
      float sj = sRaw[j];
      if (sj > myS || (sj == myS && j < tid)) rank++;
    }
    sS[rank] = myS;
    sX1[rank] = bx1; sY1[rank] = by1; sX2[rank] = bx2; sY2[rank] = by2;
    sA[rank] = (bx2 - bx1) * (by2 - by1);
    sIdx[rank] = tid;
  }
  __syncthreads();
  if (tid < NPIX) sKeep[tid] = (sS[tid] >= CONF_T) ? 1 : 0;

  for (int i = 0; i < NPIX; ++i) {
    __syncthreads();
    if (!sKeep[i]) continue;   // uniform branch (shared value after barrier)
    float x1i = sX1[i], y1i = sY1[i], x2i = sX2[i], y2i = sY2[i], ai = sA[i];
    for (int j = tid; j < NPIX; j += 384) {
      if (j > i && sKeep[j]) {
        float ww = fmaxf(1e-28f, fminf(x2i, sX2[j]) - fmaxf(x1i, sX1[j]));
        float hh = fmaxf(1e-28f, fminf(y2i, sY2[j]) - fmaxf(y1i, sY1[j]));
        float inter = ww * hh;
        float iou = inter / (ai + sA[j] - inter + 1e-14f);
        if (iou > NMS_T) sKeep[j] = 0;
      }
    }
  }
  __syncthreads();
  if (tid < NPIX && sKeep[tid]) keepOut[b * NPIX + sIdx[tid]] = 1.0f;
}

// ---------------- launch ----------------
extern "C" void kernel_launch(void* const* d_in, const int* in_sizes, int n_in,
                              void* d_out, int out_size, void* d_ws, size_t ws_size,
                              hipStream_t stream) {
  (void)in_sizes; (void)n_in; (void)out_size; (void)ws_size;
  const float* x    = (const float*)d_in[0];
  const float* wC1  = (const float*)d_in[1];   const float* bC1  = (const float*)d_in[2];
  const float* wC2  = (const float*)d_in[3];   const float* bC2  = (const float*)d_in[4];
  const float* wR1  = (const float*)d_in[5];   const float* bR1  = (const float*)d_in[6];
  const float* wR2  = (const float*)d_in[7];   const float* bR2  = (const float*)d_in[8];
  const float* wR3  = (const float*)d_in[9];   const float* bR3  = (const float*)d_in[10];
  const float* wR4  = (const float*)d_in[11];  const float* bR4  = (const float*)d_in[12];
  const float* wObj = (const float*)d_in[13];  const float* bObj = (const float*)d_in[14];
  const float* wCls = (const float*)d_in[15];  const float* bCls = (const float*)d_in[16];
  const float* wReg = (const float*)d_in[17];  const float* bReg = (const float*)d_in[18];

  char* ws = (char*)d_ws;
  const size_t PB = (size_t)NB * PPIX * CH * sizeof(_Float16);  // 7,225,344 B per act buffer
  _Float16* actX = (_Float16*)(ws);
  _Float16* T1   = (_Float16*)(ws + PB);
  _Float16* T2   = (_Float16*)(ws + 2 * PB);
  _Float16* CF   = (_Float16*)(ws + 3 * PB);
  _Float16* WH   = (_Float16*)(ws + 4 * PB);                    // 4,718,592 B
  float* heads   = (float*)(ws + 4 * PB + (size_t)9 * CH * CH * sizeof(_Float16));
  float* dout    = (float*)d_out;

  const dim3 tb(256);
  const dim3 gConv(8, NPIX);
  const dim3 bConv(128);
  const int nX = G_TOT * CH;
  const int nW = CH * CH * 9;
  const int nZ16 = (int)(4 * PB / 16);   // zero all 4 padded act buffers (halo incl.)

  k_zero<<<dim3((nZ16 + 255) / 256), tb, 0, stream>>>((uint4*)ws, nZ16);
  k_convert_x<<<dim3((nX + 255) / 256), tb, 0, stream>>>(x, actX);

  // cls branch: x -> T1 -> CF
  k_convert_w<<<dim3((nW + 255) / 256), tb, 0, stream>>>(wC1, WH);
  k_conv3x3_wmma<<<gConv, bConv, 0, stream>>>(actX, WH, bC1, T1);
  k_convert_w<<<dim3((nW + 255) / 256), tb, 0, stream>>>(wC2, WH);
  k_conv3x3_wmma<<<gConv, bConv, 0, stream>>>(T1, WH, bC2, CF);

  // reg branch: x -> T1 -> T2 -> T1 -> T2 (rf = T2)
  k_convert_w<<<dim3((nW + 255) / 256), tb, 0, stream>>>(wR1, WH);
  k_conv3x3_wmma<<<gConv, bConv, 0, stream>>>(actX, WH, bR1, T1);
  k_convert_w<<<dim3((nW + 255) / 256), tb, 0, stream>>>(wR2, WH);
  k_conv3x3_wmma<<<gConv, bConv, 0, stream>>>(T1, WH, bR2, T2);
  k_convert_w<<<dim3((nW + 255) / 256), tb, 0, stream>>>(wR3, WH);
  k_conv3x3_wmma<<<gConv, bConv, 0, stream>>>(T2, WH, bR3, T1);
  k_convert_w<<<dim3((nW + 255) / 256), tb, 0, stream>>>(wR4, WH);
  k_conv3x3_wmma<<<gConv, bConv, 0, stream>>>(T1, WH, bR4, T2);

  k_heads<<<dim3((G_TOT * 25 + 255) / 256), tb, 0, stream>>>(
      CF, T2, wObj, bObj, wCls, bCls, wReg, bReg, heads);
  k_decode<<<dim3((G_TOT + 255) / 256), tb, 0, stream>>>(heads, dout);
  k_nms<<<dim3(NCLS, NB), dim3(384), 0, stream>>>(dout, dout + (size_t)G_TOT * 6);
}
